// ModulatedDeformXCorrDepthwise_18794776887882
// MI455X (gfx1250) — matrix operand order repaired and checked by reading, not compiled, not run on hardware
//
#include <hip/hip_runtime.h>
#include <hip/hip_bf16.h>
#include <math.h>

typedef __attribute__((ext_vector_type(16))) __bf16 v16bf;
typedef __attribute__((ext_vector_type(8)))  float  v8f;

#define Bn    8
#define Cn    256
#define Hn    31
#define Wn    31
#define Tn    7
#define KK    49
#define OCH   147      // 3*T*T output channels of offset/mask conv
#define OH    25
#define OW    25
#define HP    33       // 31 + halo of 1 on each side
#define NPIX  961      // 31*31
#define KTOT  2304     // 256*9, ordered k = kidx*256 + c
#define KSTEPS 72      // 2304/32
#define MT    10       // ceil(147/16)  -> one wave per M-tile
#define NT    61       // ceil(961/16)

#define APK_ELEMS ((size_t)MT * KSTEPS * 32 * 16)   // 368640 bf16
#define SRB_ELEMS ((size_t)Bn * HP * HP * Cn)       // 2230272 bf16
#define APK_BYTES ((APK_ELEMS * 2 + 255) & ~(size_t)255)
#define SRB_BYTES ((SRB_ELEMS * 2 + 255) & ~(size_t)255)

// ---------------------------------------------------------------------------
// Pack w_offset_mask (147,256,3,3) f32 -> bf16 A-fragments in WMMA lane order.
//   klocal(e) = ((l&16)?16:0) + e   (contiguous K order, matches B LDS read)
// K ordering: k = kidx*256 + c  (kidx = 3x3 tap, c = input channel)
// ---------------------------------------------------------------------------
__global__ void pack_a_kernel(const float* __restrict__ w, __bf16* __restrict__ apk) {
    size_t tid = (size_t)blockIdx.x * blockDim.x + threadIdx.x;
    if (tid >= APK_ELEMS) return;
    int e    = (int)(tid & 15);
    int lane = (int)((tid >> 4) & 31);
    int tile = (int)(tid >> 9);
    int kt   = tile % KSTEPS;
    int mt   = tile / KSTEPS;
    int base1 = (lane & 16) ? 16 : 0;
    int base2 = (lane & 16) ? 24 : 8;
    int klocal = (e < 8) ? (base1 + e) : (base2 + (e - 8));
    int k    = kt * 32 + klocal;
    int kidx = k >> 8;
    int c    = k & 255;
    int m    = mt * 16 + (lane & 15);
    float v = 0.0f;
    if (m < OCH) v = w[(size_t)m * KTOT + (size_t)c * 9 + kidx];
    apk[tid] = (__bf16)v;
}

// ---------------------------------------------------------------------------
// sr (8,256,31,31) f32 NCHW -> bf16 NHWC with 1-px zero halo: [8][33][33][256]
// ---------------------------------------------------------------------------
__global__ void pack_sr_kernel(const float* __restrict__ sr, __bf16* __restrict__ srb) {
    size_t tid = (size_t)blockIdx.x * blockDim.x + threadIdx.x;
    if (tid >= SRB_ELEMS) return;
    int c = (int)(tid & 255);
    size_t r = tid >> 8;
    int x = (int)(r % HP); r /= HP;
    int y = (int)(r % HP);
    int b = (int)(r / HP);
    float v = 0.0f;
    if (y >= 1 && y <= Hn && x >= 1 && x <= Wn)
        v = sr[(((size_t)b * Cn + c) * Hn + (y - 1)) * Wn + (x - 1)];
    srb[tid] = (__bf16)v;
}

// ---------------------------------------------------------------------------
// Implicit-GEMM 3x3 conv via bf16 WMMA.
// Block = 320 threads = 10 waves; block owns one (batch, 16-pixel) column tile,
// wave w owns M-tile w. Per 3x3 tap, the 16px x 256ch bf16 B-panel (8 KB) is
// staged global->LDS with async b128 copies (ASYNCcnt), double-buffered so the
// next tap's DMA overlaps the current tap's 8 WMMAs.
// ---------------------------------------------------------------------------
__global__ void __launch_bounds__(320)
conv_wmma_kernel(const __bf16* __restrict__ apk,
                 const __bf16* __restrict__ srb,
                 const float*  __restrict__ bias,
                 float*        __restrict__ om) {
    __shared__ __bf16 sB[2][16 * Cn];   // 2 x 8 KB tap panels

    const int nt = blockIdx.x % NT;
    const int b  = blockIdx.x / NT;
    const int lane = threadIdx.x & 31;
    const int mt   = threadIdx.x >> 5;          // 0..9, one wave per M-tile

    // --- async stage of one tap panel: 16 pixel-rows x 512B = 512 b128 xfers
    auto stage = [&](int tap, int buf) {
        int dy = tap / 3, dx = tap % 3;
        for (int idx = threadIdx.x; idx < 512; idx += 320) {
            int p  = idx >> 5;                  // pixel 0..15
            int ch = idx & 31;                  // 16-byte chunk 0..31
            int n = nt * 16 + p; if (n > NPIX - 1) n = NPIX - 1;
            int y = n / Wn, x = n % Wn;
            const __bf16* g = srb +
                (((size_t)(b * HP + y + dy)) * HP + (x + dx)) * Cn + ch * 8;
            unsigned lds = (unsigned)(size_t)(&sB[buf][p * Cn + ch * 8]);
            unsigned long long ga = (unsigned long long)(const void*)g;
            asm volatile("global_load_async_to_lds_b128 %0, %1, off"
                         :: "v"(lds), "v"(ga) : "memory");
        }
    };

    stage(0, 0);
    asm volatile("s_wait_asynccnt 0x0" ::: "memory");
    __syncthreads();

    const int p     = lane & 15;                 // B column (pixel in tile)
    const int chalf = (lane & 16) ? 16 : 0;      // K-half within each 32-chunk

    v8f acc = {};
    for (int tap = 0; tap < 9; ++tap) {
        if (tap + 1 < 9) stage(tap + 1, (tap + 1) & 1);

        const __bf16* bbase = &sB[tap & 1][p * Cn + chalf];
        const __bf16* abase =
            apk + ((size_t)(mt * KSTEPS + tap * 8) * 32 + lane) * 16;
        #pragma unroll
        for (int ks = 0; ks < 8; ++ks) {         // 8 x K=32 -> full 256 channels
            v16bf af = *(const v16bf*)(abase + (size_t)ks * 512);
            v16bf bf = *(const v16bf*)(bbase + ks * 32);
            acc = __builtin_amdgcn_wmma_f32_16x16x32_bf16(
                false, af, false, bf, (short)0, acc, false, false);
        }

        asm volatile("s_wait_asynccnt 0x0" ::: "memory");
        __syncthreads();                         // panel tap+1 ready; tap freed
    }

    // Store: lanes 0-15 -> M=r, N=lane; lanes 16-31 -> M=8+r, N=lane-16
    int n = nt * 16 + p;
    if (n < NPIX) {
        int mrow0 = mt * 16 + ((lane & 16) ? 8 : 0);
        #pragma unroll
        for (int r = 0; r < 8; ++r) {
            int m = mrow0 + r;
            if (m < OCH)
                om[((size_t)b * OCH + m) * NPIX + n] = acc[r] + bias[m];
        }
    }
}

// ---------------------------------------------------------------------------
// Deformable bilinear gather + depthwise xcorr.
// One block per (b,i,j); one thread per channel c (256).
// ---------------------------------------------------------------------------
__global__ void deform_xcorr_kernel(const __bf16* __restrict__ srb,
                                    const float*  __restrict__ om,
                                    const float*  __restrict__ tmpl,
                                    float*        __restrict__ out) {
    __shared__ float s_oy[KK], s_ox[KK], s_m[KK];
    int bp = blockIdx.x;
    int b   = bp / (OH * OW);
    int pos = bp % (OH * OW);
    int i = pos / OW;
    int j = pos % OW;
    int c = threadIdx.x;

    if (c < KK) {
        size_t pix = (size_t)i * Wn + j;
        s_oy[c] = om[((size_t)b * OCH + c) * NPIX + pix];
        s_ox[c] = om[((size_t)b * OCH + KK + c) * NPIX + pix];
        float mm = om[((size_t)b * OCH + 2 * KK + c) * NPIX + pix];
        s_m[c] = 1.0f / (1.0f + __expf(-mm));
    }
    __syncthreads();

    const __bf16* sb = srb + (size_t)b * HP * HP * Cn + c;
    const float* tp = tmpl + ((size_t)b * Cn + c) * KK;

    float acc = 0.0f;
    #pragma unroll 1
    for (int k = 0; k < KK; ++k) {
        int ky = k / Tn, kx = k % Tn;
        float ys = (float)(i + ky) + s_oy[k];
        float xs = (float)(j + kx) + s_ox[k];
        float y0 = floorf(ys), x0 = floorf(xs);
        float wy = ys - y0,    wx = xs - x0;

        float v = 0.0f;
        #pragma unroll
        for (int q = 0; q < 4; ++q) {
            float yf = y0 + (float)(q >> 1);
            float xf = x0 + (float)(q & 1);
            bool valid = (yf >= 0.0f) && (yf <= (float)(Hn - 1)) &&
                         (xf >= 0.0f) && (xf <= (float)(Wn - 1));
            int yc = (int)fminf(fmaxf(yf, 0.0f), (float)(Hn - 1));
            int xc = (int)fminf(fmaxf(xf, 0.0f), (float)(Wn - 1));
            float g = valid
                ? (float)sb[((size_t)(yc + 1) * HP + (xc + 1)) * Cn]
                : 0.0f;
            float wgt = ((q >> 1) ? wy : (1.0f - wy)) * ((q & 1) ? wx : (1.0f - wx));
            v = fmaf(wgt, g, v);
        }
        acc = fmaf(tp[k] * s_m[k], v, acc);
    }
    out[(((size_t)b * Cn + c) * OH + i) * OW + j] = acc;
}

// ---------------------------------------------------------------------------
extern "C" void kernel_launch(void* const* d_in, const int* in_sizes, int n_in,
                              void* d_out, int out_size, void* d_ws, size_t ws_size,
                              hipStream_t stream) {
    const float* sr   = (const float*)d_in[0];   // (8,256,31,31)
    const float* tmpl = (const float*)d_in[1];   // (8,256,7,7)
    const float* w_om = (const float*)d_in[2];   // (147,256,3,3)
    const float* b_om = (const float*)d_in[3];   // (147,)
    float* out = (float*)d_out;

    char* ws = (char*)d_ws;
    __bf16* apk = (__bf16*)ws;                           // packed A fragments
    __bf16* srb = (__bf16*)(ws + APK_BYTES);             // NHWC bf16 halo image
    float*  om  = (float*) (ws + APK_BYTES + SRB_BYTES); // conv output f32

    {   // pack weights into WMMA A-fragment order
        int blocks = (int)((APK_ELEMS + 255) / 256);
        pack_a_kernel<<<blocks, 256, 0, stream>>>(w_om, apk);
    }
    {   // NCHW f32 -> NHWC bf16 (+halo)
        int blocks = (int)((SRB_ELEMS + 255) / 256);
        pack_sr_kernel<<<blocks, 256, 0, stream>>>(sr, srb);
    }
    {   // WMMA implicit-GEMM conv: one block per (batch, 16-px tile)
        int blocks = Bn * NT;    // 488
        conv_wmma_kernel<<<blocks, 320, 0, stream>>>(apk, srb, b_om, om);
    }
    {   // deformable gather + depthwise correlation
        int blocks = Bn * OH * OW;   // 5000
        deform_xcorr_kernel<<<blocks, 256, 0, stream>>>(srb, om, tmpl, out);
    }
}
